// SAM_87892210745940
// MI455X (gfx1250) — compile-verified, matrix operand order checked by reading, not run on hardware
//
#include <hip/hip_runtime.h>

#define N_ 8
#define C_ 64
#define H_ 2048

typedef __attribute__((ext_vector_type(16))) _Float16 v16h;
typedef __attribute__((ext_vector_type(8)))  _Float16 v8h;
typedef __attribute__((ext_vector_type(8)))  float    v8f;

union Frag16 { v16h v; v8h h[2]; };

// ---------------------------------------------------------------------------
// Fragment loader for 16-bit A/B operands of v_wmma_*_16x16x32_f16.
// Layout (ISA 7.12.2): lanes 0-15 hold K {0..7, 16..23}, lanes 16-31 hold
// K {8..15, 24..31}; row/col index = lane & 15. Each lane reads two
// contiguous 16B chunks from its row.
// ---------------------------------------------------------------------------
__device__ inline v16h load_frag_h(const _Float16* rowbase, unsigned lane) {
  unsigned koff = (lane & 16u) ? 8u : 0u;
  Frag16 f;
  f.h[0] = *(const v8h*)(rowbase + koff);
  f.h[1] = *(const v8h*)(rowbase + 16 + koff);
  return f.v;
}

// Same fragment pattern, but sourced from fp32 memory with convert (weights).
__device__ inline v16h load_frag_f32(const float* rowbase, unsigned lane) {
  unsigned koff = (lane & 16u) ? 8u : 0u;
  Frag16 f;
#pragma unroll
  for (int t = 0; t < 8; ++t) f.h[0][t] = (_Float16)rowbase[koff + t];
#pragma unroll
  for (int t = 0; t < 8; ++t) f.h[1][t] = (_Float16)rowbase[16 + koff + t];
  return f.v;
}

__device__ inline v8f wmma16(v16h a, v16h b, v8f c) {
  return __builtin_amdgcn_wmma_f32_16x16x32_f16(false, a, false, b,
                                                (short)0, c, false, false);
}

// ---------------------------------------------------------------------------
// Tensor Data Mover: build a D# (groups 0-3) and issue TENSOR_LOAD_TO_LDS.
// Fields packed per CDNA5 ISA §8.3-8.6. data_size = 2 bytes.
// This toolchain's builtin takes 6 args (extra int32x8 group before cpol).
// ---------------------------------------------------------------------------
#if defined(__has_builtin)
#if __has_builtin(__builtin_amdgcn_tensor_load_to_lds)
#define USE_TDM 1
#endif
#endif
#ifndef USE_TDM
#define USE_TDM 0
#endif

#if USE_TDM
typedef __attribute__((ext_vector_type(4))) unsigned int tdm_g0_t;
typedef __attribute__((ext_vector_type(8))) int          tdm_g1_t;
typedef __attribute__((ext_vector_type(4))) int          tdm_g23_t;

__device__ inline void tdm_load_tile(unsigned lds_byte_off, const void* gptr,
                                     unsigned tensor_d0, unsigned tensor_d1,
                                     unsigned tile_d0, unsigned tile_d1,
                                     unsigned long long d0_stride_elems) {
  unsigned long long ga = (unsigned long long)(size_t)gptr;
  tdm_g0_t g0;
  g0[0] = 1u;                                            // count=1 (valid D#)
  g0[1] = lds_byte_off;                                  // LDS byte address
  g0[2] = (unsigned)ga;                                  // global_addr[31:0]
  g0[3] = (unsigned)((ga >> 32) & 0x01ffffffu) | (2u << 30); // addr[56:32]|type=2
  tdm_g1_t g1;
  g1[0] = (int)(1u << 16);                               // data_size=2B, mask=0
  g1[1] = (int)((tensor_d0 & 0xffffu) << 16);            // tensor_dim0[15:0]
  g1[2] = (int)((tensor_d0 >> 16) | ((tensor_d1 & 0xffffu) << 16));
  g1[3] = (int)((tensor_d1 >> 16) | ((tile_d0 & 0xffffu) << 16));
  g1[4] = (int)(tile_d1 & 0xffffu);                      // tile_dim1 (dim2=0)
  g1[5] = (int)(unsigned)(d0_stride_elems & 0xffffffffu);
  g1[6] = (int)(unsigned)((d0_stride_elems >> 32) & 0xffffu);
  g1[7] = 0;
  tdm_g23_t gz4 = {0, 0, 0, 0};
  tdm_g1_t  gz8 = {0, 0, 0, 0, 0, 0, 0, 0};
  __builtin_amdgcn_tensor_load_to_lds(g0, g1, gz4, gz4, gz8, 0);
}
#endif

// ---------------------------------------------------------------------------
// Kernel 1: QKV projection (1x1 conv == 64x64 channel matmul).
// Writes Q,K as (n, h, c) f16 and V as (n, c, h) f16 — layouts chosen so all
// attention WMMA fragments are per-lane contiguous loads.
// 4 waves/block; each wave owns one 16-wide h tile of one batch.
// ---------------------------------------------------------------------------
__global__ __launch_bounds__(128) void qkv_proj_kernel(
    const float* __restrict__ x,
    const float* __restrict__ Wq, const float* __restrict__ bq,
    const float* __restrict__ Wk, const float* __restrict__ bk,
    const float* __restrict__ Wv, const float* __restrict__ bv,
    _Float16* __restrict__ Qhc, _Float16* __restrict__ Khc,
    _Float16* __restrict__ Vch) {
  __shared__ _Float16 xT[4][16 * 64];   // per-wave x^T tile: [h][c] f16

  const unsigned tid = threadIdx.x;
  const unsigned wave = tid >> 5, lane = tid & 31;
  const unsigned tile = blockIdx.x * 4 + wave;   // 0..1023
  const unsigned n = tile >> 7;                  // batch
  const unsigned h0 = (tile & 127) * 16;

  // Stage x^T: xT[hh][c] = x[n, c, h0+hh]   (coalesced f32 loads, cvt to f16)
  const float* xb = x + (size_t)n * C_ * H_ + h0;
  const unsigned hh = lane & 15;
  for (unsigned c = lane >> 4; c < C_; c += 2)
    xT[wave][hh * C_ + c] = (_Float16)xb[(size_t)c * H_ + hh];
  // same-wave LDS ops are in-order; no cross-wave sharing of this region

  // B fragments of x^T (K dim = c), shared by all three projections
  const _Float16* xrow = &xT[wave][(lane & 15) * C_];
  v16h bx0 = load_frag_h(xrow, lane);
  v16h bx1 = load_frag_h(xrow + 32, lane);

  const unsigned col = lane & 15;
  const unsigned halfoff = (lane >> 4) << 3;     // 0 or 8 (C/D row offset)

  const float* Ws[3] = {Wq, Wk, Wv};
  const float* bs[3] = {bq, bk, bv};
#pragma unroll
  for (int p = 0; p < 3; ++p) {
#pragma unroll
    for (int ot = 0; ot < 4; ++ot) {
      const float* wrow = Ws[p] + (ot * 16 + (lane & 15)) * C_;
      v16h a0 = load_frag_f32(wrow, lane);        // A[o, c] chunk c 0..31
      v16h a1 = load_frag_f32(wrow + 32, lane);   // chunk c 32..63
      v8f acc = {};
      acc = wmma16(a0, bx0, acc);
      acc = wmma16(a1, bx1, acc);
      if (p < 2) {
        // Q/K stored (h, c): per-lane g is contiguous in c -> one 16B store
        _Float16* dst = (p == 0) ? Qhc : Khc;
        v8h pk;
#pragma unroll
        for (int g = 0; g < 8; ++g)
          pk[g] = (_Float16)(acc[g] + bs[p][ot * 16 + halfoff + g]);
        *(v8h*)(dst + ((size_t)n * H_ + h0 + col) * C_ + ot * 16 + halfoff) = pk;
      } else {
        // V stored (c, h): scattered 2B stores (32B segments across lanes)
#pragma unroll
        for (int g = 0; g < 8; ++g) {
          unsigned o = ot * 16 + halfoff + g;
          Vch[((size_t)n * C_ + o) * H_ + h0 + col] =
              (_Float16)(acc[g] + bs[2][o]);
        }
      }
    }
  }
}

// ---------------------------------------------------------------------------
// Kernel 2: fused flash attention, S^T formulation, double-buffered TDM.
//   S^T[j,i] = sum_c K[j,c] Q[i,c]   (A = K rows, B = Q cols)
// C/D column index i == lane&15  ->  softmax stats are per-lane scalars.
// C/D row layout of S^T (j = g + 8*lanehalf) is lane-identical to the
// 16-bit B-fragment K layout, so exp(S^T) packs into the PV operand with
// per-lane cvt only.  O^T[c,i] += V[c,j] * P^T[j,i].
// Pipeline: wave 0 issues TDM loads for tile t+1 into buf^1, then waits
// TENSORcnt<=2 (tile t, issued last iteration, completes in-order) so HBM
// latency overlaps the 16 WMMAs of tile t.
// ---------------------------------------------------------------------------
__global__ __launch_bounds__(128) void attn_kernel(
    const _Float16* __restrict__ Qhc, const _Float16* __restrict__ Khc,
    const _Float16* __restrict__ Vch, const float* __restrict__ x,
    const float* __restrict__ lam, float* __restrict__ out) {
  __shared__ _Float16 Kt[2][64 * 64];  // (j, c) row-major, contiguous tiles
  __shared__ _Float16 Vt[2][64 * 64];  // (c, j) row-major, strided tiles

  const unsigned tid = threadIdx.x, wave = tid >> 5, lane = tid & 31;
  const unsigned n = blockIdx.x >> 5;                    // 32 i-blocks per n
  const unsigned i0 = (blockIdx.x & 31) * 64 + wave * 16;
  const unsigned col = lane & 15;                        // i within tile

  // Q as B fragments: B[c, i] = Q[i, c]; lane reads its own (h,c) row
  const _Float16* qrow = Qhc + ((size_t)n * H_ + i0 + col) * C_;
  v16h bq0 = load_frag_h(qrow, lane);
  v16h bq1 = load_frag_h(qrow + 32, lane);

  v8f o_acc[4] = {};                 // O^T frags: c-subtiles 0..3, col = i
  float m = -3e38f, l = 0.0f;

  const _Float16* Kbase = Khc + (size_t)n * H_ * C_;  // tile j0: + j0*C_
  const _Float16* Vbase = Vch + (size_t)n * C_ * H_;  // tile j0: + j0

#if USE_TDM
  const unsigned ldsK0 = (unsigned)(size_t)&Kt[0][0];
  const unsigned ldsK1 = (unsigned)(size_t)&Kt[1][0];
  const unsigned ldsV0 = (unsigned)(size_t)&Vt[0][0];
  const unsigned ldsV1 = (unsigned)(size_t)&Vt[1][0];
  // Prime the pipeline: tile 0 into buffer 0.
  if (wave == 0) {
    tdm_load_tile(ldsK0, Kbase, 4096u, 1u, 4096u, 0u, 4096ull);
    tdm_load_tile(ldsV0, Vbase, (unsigned)H_, 64u, 64u, 64u,
                  (unsigned long long)H_);
  }
#endif

  for (unsigned it = 0; it < H_ / 64; ++it) {
    const unsigned cur = it & 1u;
    __syncthreads();   // prev compute done; buf cur^1 free for overwrite
#if USE_TDM
    if (wave == 0) {
      if (it + 1 < H_ / 64) {
        const unsigned jn = (it + 1) * 64;
        tdm_load_tile(cur ? ldsK0 : ldsK1, Kbase + (size_t)jn * C_,
                      4096u, 1u, 4096u, 0u, 4096ull);
        tdm_load_tile(cur ? ldsV0 : ldsV1, Vbase + jn,
                      (unsigned)H_, 64u, 64u, 64u, (unsigned long long)H_);
        __builtin_amdgcn_s_wait_tensorcnt(2);  // tile `it` done (in-order)
      } else {
        __builtin_amdgcn_s_wait_tensorcnt(0);
      }
    }
#else
    {
      const unsigned j0 = it * 64;
      const unsigned* ks = (const unsigned*)(Kbase + (size_t)j0 * C_);
      unsigned* kd = (unsigned*)&Kt[cur][0];
      for (unsigned t = tid; t < 2048; t += 128) kd[t] = ks[t];
      unsigned* vd = (unsigned*)&Vt[cur][0];
      for (unsigned t = tid; t < 2048; t += 128) {
        unsigned c = t >> 5, jd = t & 31;
        vd[t] = ((const unsigned*)(Vbase + (size_t)c * H_ + j0))[jd];
      }
    }
#endif
    __syncthreads();                 // tile `it` visible to all waves

    // ---- S^T = K_tile x Q : 4 j-subtiles x (2 c-chunks) WMMAs ----
    v8f s[4];
#pragma unroll
    for (int js = 0; js < 4; ++js) {
      const _Float16* krow = &Kt[cur][(js * 16 + col) * 64];
      v16h ak0 = load_frag_h(krow, lane);
      v16h ak1 = load_frag_h(krow + 32, lane);
      v8f acc = {};
      acc = wmma16(ak0, bq0, acc);
      acc = wmma16(ak1, bq1, acc);
      s[js] = acc;
    }

    // ---- online softmax over j (per-lane: column i fixed) ----
    float mt = s[0][0];
#pragma unroll
    for (int js = 0; js < 4; ++js)
#pragma unroll
      for (int g = 0; g < 8; ++g) mt = fmaxf(mt, s[js][g]);
    mt = fmaxf(mt, __shfl_xor(mt, 16, 32));   // merge lane-halves (j +8 rows)
    float m_new = fmaxf(m, mt);
    float alpha = __expf(m - m_new);

    float rs = 0.0f;
    Frag16 p01, p23;                 // P^T packed as PV B-fragments
#pragma unroll
    for (int g = 0; g < 8; ++g) {
      float e0 = __expf(s[0][g] - m_new); rs += e0;
      float e1 = __expf(s[1][g] - m_new); rs += e1;
      float e2 = __expf(s[2][g] - m_new); rs += e2;
      float e3 = __expf(s[3][g] - m_new); rs += e3;
      p01.h[0][g] = (_Float16)e0;    // K(j) = koff + g        (j 0..15)
      p01.h[1][g] = (_Float16)e1;    // K(j) = 16 + koff + g   (j 16..31)
      p23.h[0][g] = (_Float16)e2;    // j 32..47
      p23.h[1][g] = (_Float16)e3;    // j 48..63
    }
    rs += __shfl_xor(rs, 16, 32);
    l = l * alpha + rs;
    m = m_new;

    // ---- rescale O and accumulate P^T x V : 4 c-subtiles x 2 WMMAs ----
#pragma unroll
    for (int cs = 0; cs < 4; ++cs) {
#pragma unroll
      for (int g = 0; g < 8; ++g) o_acc[cs][g] *= alpha;
      const _Float16* vrow = &Vt[cur][(cs * 16 + col) * 64];
      v16h av0 = load_frag_h(vrow, lane);        // A[c, j] chunk j 0..31
      v16h av1 = load_frag_h(vrow + 32, lane);   // chunk j 32..63
      o_acc[cs] = wmma16(av0, p01.v, o_acc[cs]);
      o_acc[cs] = wmma16(av1, p23.v, o_acc[cs]);
    }
  }

  // ---- epilogue: out[n,c,i] = lam * (O/l) + x[n,c,i] ----
  const float lamv = lam[0];
  const float inv_l = 1.0f / l;
  const unsigned halfoff = (lane >> 4) << 3;
#pragma unroll
  for (int cs = 0; cs < 4; ++cs) {
#pragma unroll
    for (int g = 0; g < 8; ++g) {
      unsigned c = cs * 16 + halfoff + g;
      size_t idx = ((size_t)n * C_ + c) * H_ + i0 + col;
      out[idx] = lamv * (o_acc[cs][g] * inv_l) + x[idx];
    }
  }
}

// ---------------------------------------------------------------------------
extern "C" void kernel_launch(void* const* d_in, const int* in_sizes, int n_in,
                              void* d_out, int out_size, void* d_ws,
                              size_t ws_size, hipStream_t stream) {
  (void)in_sizes; (void)n_in; (void)out_size; (void)ws_size;
  const float* x  = (const float*)d_in[0];
  const float* Wq = (const float*)d_in[1];
  const float* bq = (const float*)d_in[2];
  const float* Wk = (const float*)d_in[3];
  const float* bk = (const float*)d_in[4];
  const float* Wv = (const float*)d_in[5];
  const float* bv = (const float*)d_in[6];
  const float* lam = (const float*)d_in[7];
  float* out = (float*)d_out;

  _Float16* Q = (_Float16*)d_ws;                    // (n, h, c) f16, 2MB
  _Float16* K = Q + (size_t)N_ * H_ * C_;           // (n, h, c) f16, 2MB
  _Float16* V = K + (size_t)N_ * H_ * C_;           // (n, c, h) f16, 2MB

  qkv_proj_kernel<<<256, 128, 0, stream>>>(x, Wq, bq, Wk, bk, Wv, bv, Q, K, V);
  attn_kernel<<<256, 128, 0, stream>>>(Q, K, V, x, lam, out);
}
